// Global_Fea_Extractor_37855841747278
// MI455X (gfx1250) — matrix operand order repaired and checked by reading, not compiled, not run on hardware
//
#include <hip/hip_runtime.h>
#include <hip/hip_bf16.h>
#include <cstddef>

// Problem constants (from reference setup): B=2, N=8192, C=128.
#define BATCH   2
#define NROWS   8192
#define CH      128
#define TILE_M  128      // output rows per workgroup (4 waves x 32 rows)
#define KCHUNK  32       // K per WMMA step (f16 K=32)
#define KSPLIT  4        // K-dimension split across workgroups
#define NTILES  (NROWS / TILE_M)   // 64
#define ROWSTRIDE 20     // LDS row stride in dwords for f16 tile (80B, 16B-aligned)

typedef _Float16 v16h __attribute__((ext_vector_type(16)));
typedef _Float16 v8h  __attribute__((ext_vector_type(8)));
typedef float    v8f  __attribute__((ext_vector_type(8)));

// ---------------------------------------------------------------------------
// Kernel 1: l1/l2 latents. One wave (32 lanes) per row: 128-ch dot product,
// scalar BatchNorm (eval), ReLU.
// ---------------------------------------------------------------------------
__global__ __launch_bounds__(256) void latents_kernel(
    const float* __restrict__ feat,
    const float* __restrict__ w1, const float* b1, const float* g1,
    const float* beta1, const float* rm1, const float* rv1,
    const float* __restrict__ w2, const float* b2, const float* g2,
    const float* beta2, const float* rm2, const float* rv2,
    float* __restrict__ l1, float* __restrict__ l2, int total_rows) {
  int wave = (int)((blockIdx.x * blockDim.x + threadIdx.x) >> 5);
  int lane = threadIdx.x & 31;
  if (wave >= total_rows) return;

  float4 f  = reinterpret_cast<const float4*>(feat + (size_t)wave * CH)[lane];
  float4 a1 = reinterpret_cast<const float4*>(w1)[lane];
  float4 a2 = reinterpret_cast<const float4*>(w2)[lane];
  float d1 = f.x * a1.x + f.y * a1.y + f.z * a1.z + f.w * a1.w;
  float d2 = f.x * a2.x + f.y * a2.y + f.z * a2.z + f.w * a2.w;
#pragma unroll
  for (int off = 16; off > 0; off >>= 1) {
    d1 += __shfl_down(d1, off, 32);
    d2 += __shfl_down(d2, off, 32);
  }
  if (lane == 0) {
    float x1 = d1 + b1[0];
    x1 = (x1 - rm1[0]) * rsqrtf(rv1[0] + 1e-5f) * g1[0] + beta1[0];
    l1[wave] = fmaxf(x1, 0.0f);
    float x2 = d2 + b2[0];
    x2 = (x2 - rm2[0]) * rsqrtf(rv2[0] + 1e-5f) * g2[0] + beta2[0];
    l2[wave] = fmaxf(x2, 0.0f);
  }
}

// ---------------------------------------------------------------------------
// Kernel 2: per-batch M = max(l1) * max(l2)  (valid since l1,l2 >= 0).
// ---------------------------------------------------------------------------
__global__ __launch_bounds__(256) void maxred_kernel(
    const float* __restrict__ l1, const float* __restrict__ l2,
    float* __restrict__ Mb, int n) {
  __shared__ float s1[256], s2[256];
  int b = blockIdx.x, tid = threadIdx.x;
  float m1 = 0.0f, m2 = 0.0f;   // ReLU outputs are >= 0
  for (int i = tid; i < n; i += 256) {
    m1 = fmaxf(m1, l1[(size_t)b * n + i]);
    m2 = fmaxf(m2, l2[(size_t)b * n + i]);
  }
  s1[tid] = m1; s2[tid] = m2;
  __syncthreads();
  for (int off = 128; off > 0; off >>= 1) {
    if (tid < off) {
      s1[tid] = fmaxf(s1[tid], s1[tid + off]);
      s2[tid] = fmaxf(s2[tid], s2[tid + off]);
    }
    __syncthreads();
  }
  if (tid == 0) Mb[b] = s1[0] * s2[0];
}

// ---------------------------------------------------------------------------
// Kernel 3: fused flash-style attention GEMM.
// Workgroup = (batch, 128-row tile, K-split). 4 waves; wave w owns rows
// [w*32, w*32+32) x all 128 channels = 2 A-fragments, 16 WMMA accumulators.
// Feature K-chunk staged in LDS pre-converted to f16 (packed k-pairs,
// [c][k] layout, 80B row stride) so each B fragment is two ds_load_b128
// feeding TWO wmmas. P tile exp() goes straight into the A-fragment layout.
// ---------------------------------------------------------------------------
__global__ __launch_bounds__(128) void attn_gemm_kernel(
    const float* __restrict__ feat, const float* __restrict__ l1g,
    const float* __restrict__ l2g, const float* __restrict__ Mb,
    float* __restrict__ part, float* __restrict__ Spart) {
  __shared__ unsigned featH[CH * ROWSTRIDE];  // f16 pairs: [c][k/2], pad 20
  __shared__ float l1s[TILE_M];
  __shared__ float l2s[KCHUNK];
  __shared__ float red[4];

  int bid  = blockIdx.x;
  int b    = bid / (NTILES * KSPLIT);
  int rem  = bid % (NTILES * KSPLIT);
  int tile = rem / KSPLIT;
  int ks   = rem % KSPLIT;

  int tid  = threadIdx.x;
  int lane = tid & 31;
  int w    = tid >> 5;        // wave 0..3
  int half = lane >> 4;       // 0: lanes 0-15, 1: lanes 16-31
  int m    = lane & 15;

  int i0   = tile * TILE_M;
  int klen = NROWS / KSPLIT;  // 2048
  int k0   = ks * klen;

  if (tid < TILE_M) l1s[tid] = l1g[(size_t)b * NROWS + i0 + tid];
  float Mv = Mb[b];

  v8f zero = {};
  v8f acc0[8], acc1[8];
#pragma unroll
  for (int c = 0; c < 8; ++c) { acc0[c] = zero; acc1[c] = zero; }

  float psum = 0.0f;
  const size_t fbase = (size_t)b * NROWS * CH;

  for (int kc = k0; kc < k0 + klen; kc += KCHUNK) {
    __syncthreads();
    // Stage feature[kc..kc+31][0..127] as f16 pairs along k.
    // 512 units of (k-pair, 4 channels); 128 threads -> 4 units each.
    // Wave v covers k-pair rows with fully-coalesced float4 loads.
#pragma unroll
    for (int q = 0; q < 4; ++q) {
      int u  = q * 128 + tid;        // 0..511
      int kk = u >> 5;               // k-pair index 0..15
      int c4 = (u & 31) * 4;         // channel group
      const float* gp = feat + fbase + (size_t)(kc + 2 * kk) * CH + c4;
      float4 v0 = *reinterpret_cast<const float4*>(gp);
      float4 v1 = *reinterpret_cast<const float4*>(gp + CH);
      union { _Float16 h[2]; unsigned u32; } p;
      p.h[0] = (_Float16)v0.x; p.h[1] = (_Float16)v1.x;
      featH[(c4 + 0) * ROWSTRIDE + kk] = p.u32;
      p.h[0] = (_Float16)v0.y; p.h[1] = (_Float16)v1.y;
      featH[(c4 + 1) * ROWSTRIDE + kk] = p.u32;
      p.h[0] = (_Float16)v0.z; p.h[1] = (_Float16)v1.z;
      featH[(c4 + 2) * ROWSTRIDE + kk] = p.u32;
      p.h[0] = (_Float16)v0.w; p.h[1] = (_Float16)v1.w;
      featH[(c4 + 3) * ROWSTRIDE + kk] = p.u32;
    }
    if (tid < KCHUNK) l2s[tid] = l2g[(size_t)b * NROWS + kc + tid];
    __syncthreads();

    // Two A fragments: rows w*32..+15 and w*32+16..+31. Contiguous k order
    // (same as B): lanes 0-15 K{0..15}, lanes 16-31 K{16..31}.
    float l1v0 = l1s[w * 32 + m];
    float l1v1 = l1s[w * 32 + 16 + m];
    int koffA = half * 16;
    v16h a0, a1;
#pragma unroll
    for (int e = 0; e < 16; ++e) {
      int k = koffA + e;
      float lv = l2s[k];
      float p0 = __expf(fmaf(l1v0, lv, -Mv));
      float p1 = __expf(fmaf(l1v1, lv, -Mv));
      psum += p0 + p1;               // each P element held exactly once
      a0[e] = (_Float16)p0;
      a1[e] = (_Float16)p1;
    }

    // B fragments: 32x16 f16; lanes 0-15 K=0-15, lanes 16-31 K=16-31,
    // N = lane%16. Contiguous 16 halves -> two ds_load_b128, feeds 2 wmmas.
#pragma unroll
    for (int ct = 0; ct < 8; ++ct) {
      const _Float16* rowp =
          reinterpret_cast<const _Float16*>(&featH[(ct * 16 + m) * ROWSTRIDE]);
      v8h lo = *reinterpret_cast<const v8h*>(rowp + half * 16);
      v8h hi = *reinterpret_cast<const v8h*>(rowp + half * 16 + 8);
      v16h bf = __builtin_shufflevector(lo, hi, 0, 1, 2, 3, 4, 5, 6, 7,
                                        8, 9, 10, 11, 12, 13, 14, 15);
      acc0[ct] = __builtin_amdgcn_wmma_f32_16x16x32_f16(
          false, a0, false, bf, (short)0, acc0[ct], false, false);
      acc1[ct] = __builtin_amdgcn_wmma_f32_16x16x32_f16(
          false, a1, false, bf, (short)0, acc1[ct], false, false);
    }
  }

  // Store fp32 partials: part[ks][b][i][c]. D layout: VGPR r -> M=r (+8 for
  // upper half-lanes), N = lane%16.
  float* pbase = part + (((size_t)ks * BATCH + b) * NROWS) * CH;
#pragma unroll
  for (int ct = 0; ct < 8; ++ct) {
    int c = ct * 16 + m;
#pragma unroll
    for (int r = 0; r < 8; ++r) {
      int i = i0 + w * 32 + r + half * 8;
      pbase[(size_t)i * CH + c] = acc0[ct][r];
      pbase[(size_t)(i + 16) * CH + c] = acc1[ct][r];
    }
  }

  // Deterministic exp-sum reduction -> one value per workgroup.
#pragma unroll
  for (int off = 16; off > 0; off >>= 1) psum += __shfl_down(psum, off, 32);
  if (lane == 0) red[w] = psum;
  __syncthreads();
  if (tid == 0) {
    float s = 0.0f;
#pragma unroll
    for (int i = 0; i < 4; ++i) s += red[i];
    Spart[bid] = s;
  }
}

// ---------------------------------------------------------------------------
// Kernel 4: S[b] = sum of the 256 per-WG partial sums of batch b; store 1/S.
// ---------------------------------------------------------------------------
__global__ __launch_bounds__(256) void ssum_kernel(
    const float* __restrict__ Spart, float* __restrict__ Sinv) {
  __shared__ float s[256];
  int b = blockIdx.x, tid = threadIdx.x;
  s[tid] = Spart[b * (NTILES * KSPLIT) + tid];   // exactly 256 per batch
  __syncthreads();
  for (int off = 128; off > 0; off >>= 1) {
    if (tid < off) s[tid] += s[tid + off];
    __syncthreads();
  }
  if (tid == 0) Sinv[b] = 1.0f / s[0];
}

// ---------------------------------------------------------------------------
// Kernel 5: out = (sum of KSPLIT partials) * (1/S) + feature.
// ---------------------------------------------------------------------------
__global__ __launch_bounds__(256) void finalize_kernel(
    const float* __restrict__ part, const float* __restrict__ Sinv,
    const float* __restrict__ feat, float* __restrict__ out) {
  size_t idx = (size_t)blockIdx.x * 256 + threadIdx.x;
  const size_t total = (size_t)BATCH * NROWS * CH;
  if (idx >= total) return;
  int b = (int)(idx / ((size_t)NROWS * CH));
  const size_t stride = (size_t)BATCH * NROWS * CH;
  float s = 0.0f;
#pragma unroll
  for (int ks = 0; ks < KSPLIT; ++ks) s += part[idx + ks * stride];
  out[idx] = s * Sinv[b] + feat[idx];
}

// ---------------------------------------------------------------------------
extern "C" void kernel_launch(void* const* d_in, const int* in_sizes, int n_in,
                              void* d_out, int out_size, void* d_ws, size_t ws_size,
                              hipStream_t stream) {
  (void)in_sizes; (void)n_in; (void)out_size; (void)ws_size;
  const float* feat  = (const float*)d_in[0];
  const float* w1    = (const float*)d_in[1];
  const float* b1    = (const float*)d_in[2];
  const float* g1    = (const float*)d_in[3];
  const float* beta1 = (const float*)d_in[4];
  const float* rm1   = (const float*)d_in[5];
  const float* rv1   = (const float*)d_in[6];
  const float* w2    = (const float*)d_in[7];
  const float* b2    = (const float*)d_in[8];
  const float* g2    = (const float*)d_in[9];
  const float* beta2 = (const float*)d_in[10];
  const float* rm2   = (const float*)d_in[11];
  const float* rv2   = (const float*)d_in[12];

  // Workspace layout (floats):
  float* ws    = (float*)d_ws;
  float* l1    = ws;                       // B*N
  float* l2    = l1 + BATCH * NROWS;       // B*N
  float* Mb    = l2 + BATCH * NROWS;       // B
  float* Sinv  = Mb + BATCH;               // B
  float* Spart = Sinv + BATCH;             // B*NTILES*KSPLIT = 512
  float* part  = Spart + BATCH * NTILES * KSPLIT;  // KSPLIT*B*N*C (32 MB)

  const int total_rows = BATCH * NROWS;    // 16384

  latents_kernel<<<total_rows / 8, 256, 0, stream>>>(
      feat, w1, b1, g1, beta1, rm1, rv1, w2, b2, g2, beta2, rm2, rv2,
      l1, l2, total_rows);

  maxred_kernel<<<BATCH, 256, 0, stream>>>(l1, l2, Mb, NROWS);

  attn_gemm_kernel<<<BATCH * NTILES * KSPLIT, 128, 0, stream>>>(
      feat, l1, l2, Mb, part, Spart);

  ssum_kernel<<<BATCH, 256, 0, stream>>>(Spart, Sinv);

  finalize_kernel<<<(BATCH * NROWS * CH) / 256, 256, 0, stream>>>(
      part, Sinv, feat, (float*)d_out);
}
